// MultiHeadSelfAttention_86028194939161
// MI455X (gfx1250) — matrix operand order repaired and checked by reading, not compiled, not run on hardware
//
#include <hip/hip_runtime.h>
#include <cstdint>

// ---------------- problem constants ----------------
#define BATCH   4
#define SEQ     2048
#define DMODEL  1024
#define NHEAD   16
#define DK      64
#define BN      (BATCH * SEQ)        // 8192 total rows
#define LOG_THETA 9.210340371976184f // ln(10000)
#define NEG_BIG  (-1.0e30f)

// ---------------- vector types ----------------
typedef __attribute__((ext_vector_type(16))) __bf16          bf16x16;
typedef __attribute__((ext_vector_type(8)))  float           f32x8;
typedef __attribute__((ext_vector_type(4)))  unsigned int    u32x4;
typedef __attribute__((ext_vector_type(8)))  unsigned short  u16x8;

// ---------------- helpers ----------------
static __device__ __forceinline__ unsigned short f2bf(float f) {
  unsigned int x = __float_as_uint(f);
  x += 0x7fffu + ((x >> 16) & 1u);   // round-to-nearest-even
  return (unsigned short)(x >> 16);
}

static __device__ __forceinline__ f32x8 wmma_bf16(bf16x16 a, bf16x16 b, f32x8 c) {
  // (neg_a, A, neg_b, B, c_mod, C, reuse_a, reuse_b)
  return __builtin_amdgcn_wmma_f32_16x16x32_bf16(false, a, false, b, (short)0, c,
                                                 false, false);
}

// A-fragment (16x32, M=lane&15) from contiguous bf16 row; chunks at K=8h and 16+8h.
static __device__ __forceinline__ bf16x16 load_a_bf16(const unsigned short* row, int half) {
  union { u32x4 q[2]; bf16x16 v; } u;
  u.q[0] = *(const u32x4*)(row + 8 * half);
  u.q[1] = *(const u32x4*)(row + 16 + 8 * half);
  return u.v;
}

// B-fragment (32x16, N=lane&15) from contiguous bf16 row; chunks at K=8h and 16+8h.
static __device__ __forceinline__ bf16x16 load_b_bf16(const unsigned short* row, int half) {
  union { u32x4 q[2]; bf16x16 v; } u;
  const u32x4* p = (const u32x4*)(row + 8 * half);
  u.q[0] = p[0];
  u.q[1] = p[2];
  return u.v;
}

// =====================================================================
// Kernel 0: one-time f32 -> bf16 conversion (bandwidth bound, ~48MB read)
// =====================================================================
__global__ __launch_bounds__(256) void cvt_bf16_kernel(
    const float* __restrict__ src, unsigned short* __restrict__ dst, int n8) {
  const int stride = gridDim.x * blockDim.x;
  for (int i = blockIdx.x * blockDim.x + threadIdx.x; i < n8; i += stride) {
    const float4* s = (const float4*)(src + (size_t)i * 8);
    float4 a = s[0], b = s[1];
    u16x8 o;
    o[0] = f2bf(a.x); o[1] = f2bf(a.y); o[2] = f2bf(a.z); o[3] = f2bf(a.w);
    o[4] = f2bf(b.x); o[5] = f2bf(b.y); o[6] = f2bf(b.z); o[7] = f2bf(b.w);
    *(u16x8*)(dst + (size_t)i * 8) = o;
  }
}

// =====================================================================
// Ping-pong GEMM core: acc[4] += A(16xD) x B(Dx64) for one 16-row tile,
// with fragments double-buffered so loads for the next 32-K step are in
// flight during the current step's WMMAs -- no loop-carried reg copies.
// =====================================================================
static __device__ __forceinline__ void gemm_16x64(
    const unsigned short* __restrict__ arow,   // A row base (lane's row)
    const unsigned short* __restrict__ wrow0,  // B row base (lane's col, j=0)
    int half, f32x8 acc[4]) {
  bf16x16 a0 = load_a_bf16(arow, half);
  bf16x16 b0[4];
#pragma unroll
  for (int j = 0; j < 4; ++j)
    b0[j] = load_b_bf16(wrow0 + (size_t)j * 16 * DMODEL, half);

  for (int kb = 0; kb < DMODEL; kb += 64) {
    // phase 1: issue loads for kb+32, compute on buf0 (kb)
    const int k1 = kb + 32;  // always < DMODEL (kb <= DMODEL-64)
    bf16x16 a1 = load_a_bf16(arow + k1, half);
    bf16x16 b1[4];
#pragma unroll
    for (int j = 0; j < 4; ++j)
      b1[j] = load_b_bf16(wrow0 + (size_t)j * 16 * DMODEL + k1, half);
#pragma unroll
    for (int j = 0; j < 4; ++j)
      acc[j] = wmma_bf16(a0, b0[j], acc[j]);

    // phase 2: issue loads for kb+64 (clamped), compute on buf1 (kb+32)
    const int k2 = (kb + 64 < DMODEL) ? (kb + 64) : 0;
    a0 = load_a_bf16(arow + k2, half);
#pragma unroll
    for (int j = 0; j < 4; ++j)
      b0[j] = load_b_bf16(wrow0 + (size_t)j * 16 * DMODEL + k2, half);
#pragma unroll
    for (int j = 0; j < 4; ++j)
      acc[j] = wmma_bf16(a1, b1[j], acc[j]);
  }
}

// =====================================================================
// Kernel 1: fused QKV projection + RoPE (+1/sqrt(dk) into Q), bf16 inputs.
//   grid(BN/16, 6), block 256 (8 waves). Wave g = by*8+w picks
//   which in {Q,K,V} (g>>4) and head (g&15); computes a 16x64 tile.
// =====================================================================
__global__ __launch_bounds__(256) void qkv_rope_kernel(
    const unsigned short* __restrict__ xb, const unsigned short* __restrict__ wqb,
    const unsigned short* __restrict__ wkb, const unsigned short* __restrict__ wvb,
    unsigned short* __restrict__ Qws, unsigned short* __restrict__ Kws,
    unsigned short* __restrict__ Vt) {
  const int lane  = threadIdx.x & 31;
  const int wave  = threadIdx.x >> 5;
  const int g     = blockIdx.y * 8 + wave;   // 0..47
  const int which = g >> 4;                  // 0=Q 1=K 2=V
  const int head  = g & 15;
  const int m0    = blockIdx.x * 16;         // global row tile in [0,8192)
  const int mlane = lane & 15;
  const int half  = lane >> 4;

  const unsigned short* W = (which == 0) ? wqb : ((which == 1) ? wkb : wvb);

  f32x8 acc[4];
  const f32x8 zero = {0, 0, 0, 0, 0, 0, 0, 0};
#pragma unroll
  for (int j = 0; j < 4; ++j) acc[j] = zero;

  const unsigned short* arow  = xb + (size_t)(m0 + mlane) * DMODEL;
  const unsigned short* wrow0 = W + (size_t)(head * DK + mlane) * DMODEL;

  gemm_16x64(arow, wrow0, half, acc);

  const int b       = m0 >> 11;          // / SEQ
  const int posbase = m0 & (SEQ - 1);
  const size_t bh   = (size_t)b * NHEAD + head;

  if (which < 2) {
    // RoPE in C-layout: lane col dd = j*16+mlane within head; pair at lane^1.
#pragma unroll
    for (int j = 0; j < 4; ++j) {
      const int dd = j * 16 + mlane;
      const int i  = dd >> 1;
      const float invf = __expf(-(float)(2 * i) * (LOG_THETA / (float)DK));
      const bool even = (dd & 1) == 0;
#pragma unroll
      for (int r = 0; r < 8; ++r) {
        const float pos = (float)(posbase + r + 8 * half);
        const float ang = pos * invf;
        const float cs = __cosf(ang), sn = __sinf(ang);
        const float v = acc[j][r];
        const float partner = __shfl_xor(v, 1);
        acc[j][r] = even ? (cs * v - sn * partner) : (sn * partner + cs * v);
      }
    }
    const float scale = (which == 0) ? 0.125f : 1.0f;  // 1/sqrt(64) folded into Q
    unsigned short* dst = (which == 0) ? Qws : Kws;
#pragma unroll
    for (int j = 0; j < 4; ++j)
#pragma unroll
      for (int r = 0; r < 8; ++r) {
        const size_t idx = (bh * SEQ + posbase + r + 8 * half) * DK + j * 16 + mlane;
        dst[idx] = f2bf(acc[j][r] * scale);
      }
  } else {
    // V stored transposed: Vt[b,h,dk,n]
#pragma unroll
    for (int j = 0; j < 4; ++j)
#pragma unroll
      for (int r = 0; r < 8; ++r) {
        const size_t idx = (bh * DK + j * 16 + mlane) * (size_t)SEQ + posbase + r + 8 * half;
        Vt[idx] = f2bf(acc[j][r]);
      }
  }
}

// =====================================================================
// Kernel 2: causal flash attention.
//   grid(SEQ/128, B*H), block 256 (8 waves). Each wave owns 16 q rows,
//   iterates kv tiles of 32, online softmax in f32, S/P via bf16 WMMA.
//   All K AND V fragments issued at the top of each kv iteration so the
//   V loads hide behind the S WMMAs + softmax.
// =====================================================================
__global__ __launch_bounds__(256) void attn_kernel(
    const unsigned short* __restrict__ Qws, const unsigned short* __restrict__ Kws,
    const unsigned short* __restrict__ Vt, unsigned short* __restrict__ ctx) {
  __shared__ __align__(16) unsigned short Pbuf[8][16 * 32];

  const int lane  = threadIdx.x & 31;
  const int wave  = threadIdx.x >> 5;
  const int bh    = blockIdx.y;            // b*NHEAD + head
  const int b     = bh >> 4;
  const int head  = bh & 15;
  const int q0    = blockIdx.x * 128 + wave * 16;  // seq position of tile
  const int mlane = lane & 15;
  const int half  = lane >> 4;

  const unsigned short* Qbase = Qws + (size_t)bh * SEQ * DK;
  const unsigned short* Kbase = Kws + (size_t)bh * SEQ * DK;
  const unsigned short* Vbase = Vt  + (size_t)bh * DK * SEQ;

  const unsigned short* qrow = Qbase + (size_t)(q0 + mlane) * DK;
  const bf16x16 qa0 = load_a_bf16(qrow, half);       // dk 0..31
  const bf16x16 qa1 = load_a_bf16(qrow + 32, half);  // dk 32..63

  const f32x8 zero = {0, 0, 0, 0, 0, 0, 0, 0};
  f32x8 o[4];
#pragma unroll
  for (int j = 0; j < 4; ++j) o[j] = zero;
  float mi[8], li[8];
#pragma unroll
  for (int r = 0; r < 8; ++r) { mi[r] = NEG_BIG; li[r] = 0.0f; }

  unsigned short* P = &Pbuf[wave][0];

  for (int kv0 = 0; kv0 <= q0; kv0 += 32) {
    // ---- issue ALL fragment loads up front (K for S, V for PV) ----
    const unsigned short* kr0 = Kbase + (size_t)(kv0 + mlane) * DK;
    const unsigned short* kr1 = Kbase + (size_t)(kv0 + 16 + mlane) * DK;
    const bf16x16 k00 = load_b_bf16(kr0, half);
    const bf16x16 k01 = load_b_bf16(kr0 + 32, half);
    const bf16x16 k10 = load_b_bf16(kr1, half);
    const bf16x16 k11 = load_b_bf16(kr1 + 32, half);
    bf16x16 vf[4];
#pragma unroll
    for (int j = 0; j < 4; ++j)
      vf[j] = load_b_bf16(Vbase + (size_t)(j * 16 + mlane) * SEQ + kv0, half);

    // ---- S = Q K^T for two 16-wide kv subtiles ----
    f32x8 s0 = zero, s1 = zero;
    s0 = wmma_bf16(qa0, k00, s0);
    s0 = wmma_bf16(qa1, k01, s0);
    s1 = wmma_bf16(qa0, k10, s1);
    s1 = wmma_bf16(qa1, k11, s1);

    // ---- causal mask ----
    const int kg0 = kv0 + mlane, kg1 = kv0 + 16 + mlane;
#pragma unroll
    for (int r = 0; r < 8; ++r) {
      const int qg = q0 + r + 8 * half;
      if (kg0 > qg) s0[r] = NEG_BIG;
      if (kg1 > qg) s1[r] = NEG_BIG;
    }
    // ---- online softmax (per-row stats live in each 16-lane half) ----
#pragma unroll
    for (int r = 0; r < 8; ++r) {
      float v = fmaxf(s0[r], s1[r]);
#pragma unroll
      for (int off = 8; off >= 1; off >>= 1) v = fmaxf(v, __shfl_xor(v, off));
      const float mn = fmaxf(mi[r], v);
      const float alpha = __expf(mi[r] - mn);
      const float p0 = __expf(s0[r] - mn);
      const float p1 = __expf(s1[r] - mn);
      float sum = p0 + p1;
#pragma unroll
      for (int off = 8; off >= 1; off >>= 1) sum += __shfl_xor(sum, off);
      li[r] = li[r] * alpha + sum;
      mi[r] = mn;
#pragma unroll
      for (int j = 0; j < 4; ++j) o[j][r] *= alpha;
      // stash P (row-major 16x32 bf16) for C->A relayout
      P[(r + 8 * half) * 32 + mlane]      = f2bf(p0);
      P[(r + 8 * half) * 32 + 16 + mlane] = f2bf(p1);
    }
    // ---- O += P V ----
    const bf16x16 pa = load_a_bf16(P + mlane * 32, half);
#pragma unroll
    for (int j = 0; j < 4; ++j)
      o[j] = wmma_bf16(pa, vf[j], o[j]);
  }

  // ---- normalize + write context (bf16, [BN, DMODEL]) ----
#pragma unroll
  for (int r = 0; r < 8; ++r) {
    const float inv = 1.0f / li[r];
    const size_t rowidx = ((size_t)b * SEQ + q0 + r + 8 * half) * DMODEL + head * DK;
#pragma unroll
    for (int j = 0; j < 4; ++j)
      ctx[rowidx + j * 16 + mlane] = f2bf(o[j][r] * inv);
  }
}

// =====================================================================
// Kernel 3: output projection  out[m,e] = sum_d ctx[m,d] * Wo[e,d]
//   grid(BN/16, 2), block 256 (8 waves); wave handles a 16x64 f32 tile.
// =====================================================================
__global__ __launch_bounds__(256) void out_proj_kernel(
    const unsigned short* __restrict__ ctx, const unsigned short* __restrict__ wob,
    float* __restrict__ out) {
  const int lane  = threadIdx.x & 31;
  const int wave  = threadIdx.x >> 5;
  const int g     = blockIdx.y * 8 + wave;   // 0..15 -> 64-col group
  const int m0    = blockIdx.x * 16;
  const int mlane = lane & 15;
  const int half  = lane >> 4;
  const int colbase = g * 64;

  const f32x8 zero = {0, 0, 0, 0, 0, 0, 0, 0};
  f32x8 acc[4];
#pragma unroll
  for (int j = 0; j < 4; ++j) acc[j] = zero;

  const unsigned short* arow  = ctx + (size_t)(m0 + mlane) * DMODEL;
  const unsigned short* wrow0 = wob + (size_t)(colbase + mlane) * DMODEL;

  gemm_16x64(arow, wrow0, half, acc);

#pragma unroll
  for (int r = 0; r < 8; ++r) {
    const size_t rowidx = (size_t)(m0 + r + 8 * half) * DMODEL + colbase;
#pragma unroll
    for (int j = 0; j < 4; ++j)
      out[rowidx + j * 16 + mlane] = acc[j][r];
  }
}

// =====================================================================
// launcher
// =====================================================================
extern "C" void kernel_launch(void* const* d_in, const int* in_sizes, int n_in,
                              void* d_out, int out_size, void* d_ws, size_t ws_size,
                              hipStream_t stream) {
  const float* x  = (const float*)d_in[0];
  const float* Wq = (const float*)d_in[1];
  const float* Wk = (const float*)d_in[2];
  const float* Wv = (const float*)d_in[3];
  const float* Wo = (const float*)d_in[4];
  float* out = (float*)d_out;

  // workspace (bf16 / ushort elements):
  //   Q, K ([b,h,n,dk]), Vt ([b,h,dk,n]), ctx ([BN,D]), xb ([BN,D]), 4 weights
  const size_t elems = (size_t)BATCH * NHEAD * SEQ * DK;  // 8,388,608
  const size_t welems = (size_t)DMODEL * DMODEL;          // 1,048,576
  unsigned short* Qws = (unsigned short*)d_ws;
  unsigned short* Kws = Qws + elems;
  unsigned short* Vt  = Kws + elems;
  unsigned short* ctx = Vt + elems;
  unsigned short* xb  = ctx + elems;
  unsigned short* wqb = xb + elems;
  unsigned short* wkb = wqb + welems;
  unsigned short* wvb = wkb + welems;
  unsigned short* wob = wvb + welems;   // total ~88 MB

  (void)in_sizes; (void)n_in; (void)out_size; (void)ws_size;

  // one-time f32 -> bf16 conversion (bandwidth bound)
  cvt_bf16_kernel<<<dim3(2048), 256, 0, stream>>>(x, xb, (int)(elems / 8));
  cvt_bf16_kernel<<<dim3(512), 256, 0, stream>>>(Wq, wqb, (int)(welems / 8));
  cvt_bf16_kernel<<<dim3(512), 256, 0, stream>>>(Wk, wkb, (int)(welems / 8));
  cvt_bf16_kernel<<<dim3(512), 256, 0, stream>>>(Wv, wvb, (int)(welems / 8));
  cvt_bf16_kernel<<<dim3(512), 256, 0, stream>>>(Wo, wob, (int)(welems / 8));

  qkv_rope_kernel<<<dim3(BN / 16, 6), 256, 0, stream>>>(xb, wqb, wkb, wvb, Qws, Kws, Vt);
  attn_kernel<<<dim3(SEQ / 128, BATCH * NHEAD), 256, 0, stream>>>(Qws, Kws, Vt, ctx);
  out_proj_kernel<<<dim3(BN / 16, 2), 256, 0, stream>>>(ctx, wob, out);
}
